// Attn_Net_Gated_71786083385802
// MI455X (gfx1250) — matrix operand order repaired and checked by reading, not compiled, not run on hardware
//
#include <hip/hip_runtime.h>
#include <hip/hip_bf16.h>

// ---------------------------------------------------------------------------
// Attn_Net_Gated on MI455X (gfx1250, wave32, WMMA):
//   out0 = feature / max(||row||, 1e-12)                 [N, D] f32
//   out1 = segment_softmax(sigmoid(f@Wa^T+ba) * tanh(f@Wb^T+bb) @ Wc^T + bc)
// N = 262144, D = 1024, L = 128, 64 segments.
//
// Roofline: the normalized-feature copy (2 GiB HBM traffic @ 23.3 TB/s ~ 92us)
// is the floor. bf16 WMMA (16x16x32) keeps the 137 GFLOP GEMM pair under it.
// 32-row tiles amortize the L2-resident weight fragments over two A tiles
// (4 WMMAs per B load), halving L2 weight traffic vs a 16-row tile.
// Feature stream uses non-temporal loads/stores so it doesn't evict weights.
// ---------------------------------------------------------------------------

typedef __attribute__((ext_vector_type(16))) __bf16 v16bf;
typedef __attribute__((ext_vector_type(8)))  float  v8f;
typedef __attribute__((ext_vector_type(4)))  float  f4vec;

#define D_DIM 1024
#define L_DIM 128
#define NSEG  64
#define TILE_M 32
#define SPAD (D_DIM + 8)   // bf16 per LDS row: 2064B = 516 dwords -> 4-bank skew

__device__ __forceinline__ unsigned short f2bf(float f) {
  unsigned u = __float_as_uint(f);
  u += 0x7fffu + ((u >> 16) & 1u);     // round-to-nearest-even
  return (unsigned short)(u >> 16);
}

union BF16Frag { uint4 u[2]; v16bf v; };

// --- convert Wa, Wb (f32) -> contiguous bf16 [Wa_bf | Wb_bf] in workspace ---
__global__ __launch_bounds__(256) void convert_weights(
    const float* __restrict__ Wa, const float* __restrict__ Wb,
    unsigned short* __restrict__ wbf) {
  int i = blockIdx.x * blockDim.x + threadIdx.x;
  const int half = L_DIM * D_DIM;
  if (i < half)            wbf[i] = f2bf(Wa[i]);
  else if (i < 2 * half)   wbf[i] = f2bf(Wb[i - half]);
}

__global__ void seg_init_k(unsigned* __restrict__ segmax, float* __restrict__ segsum) {
  int i = threadIdx.x;
  if (i < NSEG) { segmax[i] = 0u; segsum[i] = 0.0f; }
}

// --- fused: row L2-norm + dual bf16 WMMA GEMM (2 M-tiles) + gated score ----
__global__ __launch_bounds__(256) void fused_norm_gemm(
    const float* __restrict__ feat,
    const unsigned short* __restrict__ Wa_bf,
    const unsigned short* __restrict__ Wb_bf,
    const float* __restrict__ ba, const float* __restrict__ bb,
    const float* __restrict__ Wc, const float* __restrict__ bc,
    float* __restrict__ out_norm, float* __restrict__ score_raw)
{
  __shared__ unsigned short sA[TILE_M * SPAD];  // 32 x 1024 bf16 (+pad) ~ 66 KB
  __shared__ float sScore[TILE_M];

  const int tid  = threadIdx.x;
  const int wave = tid >> 5;
  const int lane = tid & 31;
  const long long r0 = (long long)blockIdx.x * TILE_M;

  if (tid < TILE_M) sScore[tid] = 0.0f;

  // ---- Stage 1: each wave normalizes 4 rows, stages bf16 copy in LDS ----
  // Non-temporal: feature is read once, out_norm written once -> keep out of
  // the caches so the bf16 weights stay L2-resident.
  #pragma unroll
  for (int rr = 0; rr < 4; ++rr) {
    const int row = wave * 4 + rr;
    const f4vec* src = (const f4vec*)(feat + (r0 + row) * D_DIM);
    f4vec v[8];
    float ss = 0.0f;
    #pragma unroll
    for (int i = 0; i < 8; ++i) {
      v[i] = __builtin_nontemporal_load(src + lane + i * 32);
      ss += v[i].x * v[i].x + v[i].y * v[i].y + v[i].z * v[i].z + v[i].w * v[i].w;
    }
    #pragma unroll
    for (int m = 16; m >= 1; m >>= 1) ss += __shfl_xor(ss, m, 32);
    const float inv = 1.0f / fmaxf(sqrtf(ss), 1e-12f);
    f4vec* dst = (f4vec*)(out_norm + (r0 + row) * D_DIM);
    #pragma unroll
    for (int i = 0; i < 8; ++i) {
      f4vec o = v[i] * inv;
      __builtin_nontemporal_store(o, dst + lane + i * 32);
      uint2 p;
      p.x = (unsigned)f2bf(v[i].x) | ((unsigned)f2bf(v[i].y) << 16);
      p.y = (unsigned)f2bf(v[i].z) | ((unsigned)f2bf(v[i].w) << 16);
      *(uint2*)&sA[row * SPAD + (lane + i * 32) * 4] = p;
    }
  }
  __syncthreads();

  // ---- Stage 2: wave w owns output columns [16w, 16w+16) of L=128,
  //      two A tiles (rows 0-15 and 16-31) share every B fragment. ----
  const int colInTile = lane & 15;
  const int halfSel   = lane >> 4;       // 0: lanes 0-15, 1: lanes 16-31
  const int col       = wave * 16 + colInTile;

  v8f ca[2] = {}, cb[2] = {};
  // A (16x32 bf16): lanes 0-15 hold row=lane, K chunks [k0,k0+8)+[k0+16,k0+24);
  // lanes 16-31 hold row=lane-16, chunks shifted by 8 (per ISA 7.12.2).
  const unsigned short* aPtr0 = &sA[colInTile * SPAD + halfSel * 8];
  const unsigned short* aPtr1 = aPtr0 + 16 * SPAD;
  // B (32x16 bf16): lane holds col n, K chunks [k0+8*halfSel,+8)+[k0+16+8*halfSel,+8).
  const unsigned short* waPtr = Wa_bf + (size_t)col * D_DIM + halfSel * 8;
  const unsigned short* wbPtr = Wb_bf + (size_t)col * D_DIM + halfSel * 8;

  for (int k0 = 0; k0 < D_DIM; k0 += 32) {
    BF16Frag a0, a1, fa, fb;
    a0.u[0] = *(const uint4*)(aPtr0 + k0);
    a0.u[1] = *(const uint4*)(aPtr0 + k0 + 16);
    a1.u[0] = *(const uint4*)(aPtr1 + k0);
    a1.u[1] = *(const uint4*)(aPtr1 + k0 + 16);
    fa.u[0] = *(const uint4*)(waPtr + k0);
    fa.u[1] = *(const uint4*)(waPtr + k0 + 16);
    fb.u[0] = *(const uint4*)(wbPtr + k0);
    fb.u[1] = *(const uint4*)(wbPtr + k0 + 16);
    ca[0] = __builtin_amdgcn_wmma_f32_16x16x32_bf16(false, a0.v, false, fa.v,
                                                    (short)0, ca[0], false, false);
    cb[0] = __builtin_amdgcn_wmma_f32_16x16x32_bf16(false, a0.v, false, fb.v,
                                                    (short)0, cb[0], false, false);
    ca[1] = __builtin_amdgcn_wmma_f32_16x16x32_bf16(false, a1.v, false, fa.v,
                                                    (short)0, ca[1], false, false);
    cb[1] = __builtin_amdgcn_wmma_f32_16x16x32_bf16(false, a1.v, false, fb.v,
                                                    (short)0, cb[1], false, false);
  }

  // ---- Epilogue: gated activation, project with Wc, reduce per-row score ----
  const float bav = ba[col], bbv = bb[col], wcv = Wc[col];
  #pragma unroll
  for (int t = 0; t < 2; ++t) {
    float acc[8];
    #pragma unroll
    for (int r = 0; r < 8; ++r) {
      float a = ca[t][r] + bav;
      float b = cb[t][r] + bbv;
      float sg = 1.0f / (1.0f + __expf(-a));
      acc[r] = sg * tanhf(b) * wcv;
    }
    // C/D layout: VGPR r, lanes 0-15 -> row r, lanes 16-31 -> row r+8
    #pragma unroll
    for (int m = 1; m <= 8; m <<= 1) {
      #pragma unroll
      for (int r = 0; r < 8; ++r) acc[r] += __shfl_xor(acc[r], m, 32);
    }
    if (colInTile == 0) {
      #pragma unroll
      for (int r = 0; r < 8; ++r)
        atomicAdd(&sScore[t * 16 + halfSel * 8 + r], acc[r]);
    }
  }
  __syncthreads();
  if (tid < TILE_M) score_raw[r0 + tid] = sScore[tid] + bc[0];
}

// --- segment softmax helpers (order-preserving float<->uint key) ------------
__device__ __forceinline__ unsigned f32_to_key(float f) {
  unsigned b = __float_as_uint(f);
  return ((int)b < 0) ? ~b : (b | 0x80000000u);
}
__device__ __forceinline__ float key_to_f32(unsigned k) {
  unsigned b = (k & 0x80000000u) ? (k & 0x7fffffffu) : ~k;
  return __uint_as_float(b);
}

__global__ __launch_bounds__(256) void seg_max_k(
    const float* __restrict__ score, const int* __restrict__ batch,
    unsigned* __restrict__ segmax, int n) {
  __shared__ unsigned sm[NSEG];
  if (threadIdx.x < NSEG) sm[threadIdx.x] = 0u;
  __syncthreads();
  int i = blockIdx.x * blockDim.x + threadIdx.x;
  if (i < n) atomicMax(&sm[batch[i]], f32_to_key(score[i]));
  __syncthreads();
  if (threadIdx.x < NSEG && sm[threadIdx.x] != 0u)
    atomicMax(&segmax[threadIdx.x], sm[threadIdx.x]);
}

__global__ __launch_bounds__(256) void seg_expsum_k(
    const float* __restrict__ score, const int* __restrict__ batch,
    const unsigned* __restrict__ segmax, float* __restrict__ ex,
    float* __restrict__ segsum, int n) {
  __shared__ float ss[NSEG];
  if (threadIdx.x < NSEG) ss[threadIdx.x] = 0.0f;
  __syncthreads();
  int i = blockIdx.x * blockDim.x + threadIdx.x;
  if (i < n) {
    int b = batch[i];
    float e = __expf(score[i] - key_to_f32(segmax[b]));
    ex[i] = e;
    atomicAdd(&ss[b], e);
  }
  __syncthreads();
  if (threadIdx.x < NSEG && ss[threadIdx.x] != 0.0f)
    atomicAdd(&segsum[threadIdx.x], ss[threadIdx.x]);
}

__global__ __launch_bounds__(256) void seg_div_k(
    const float* __restrict__ ex, const int* __restrict__ batch,
    const float* __restrict__ segsum, float* __restrict__ out_score, int n) {
  int i = blockIdx.x * blockDim.x + threadIdx.x;
  if (i < n) out_score[i] = ex[i] / (segsum[batch[i]] + 1e-16f);
}

// ---------------------------------------------------------------------------
extern "C" void kernel_launch(void* const* d_in, const int* in_sizes, int n_in,
                              void* d_out, int out_size, void* d_ws, size_t ws_size,
                              hipStream_t stream) {
  // setup_inputs order: feature, batch, istrain, Wa, ba, Wb, bb, Wc, bc
  const float* feat  = (const float*)d_in[0];
  const int*   batch = (const int*)d_in[1];
  const float* Wa    = (const float*)d_in[3];
  const float* ba    = (const float*)d_in[4];
  const float* Wb    = (const float*)d_in[5];
  const float* bb    = (const float*)d_in[6];
  const float* Wc    = (const float*)d_in[7];
  const float* bc    = (const float*)d_in[8];
  const int N = in_sizes[0] / D_DIM;   // 262144

  // Workspace layout
  char* ws = (char*)d_ws;
  unsigned short* Wab = (unsigned short*)ws;                       // L*D bf16
  unsigned short* Wbb = Wab + (size_t)L_DIM * D_DIM;               // L*D bf16
  float* score_raw = (float*)(ws + 2ull * L_DIM * D_DIM * sizeof(unsigned short));
  float* ex        = score_raw + N;
  unsigned* segmax = (unsigned*)(ex + N);
  float* segsum    = (float*)(segmax + NSEG);

  float* out_norm  = (float*)d_out;
  float* out_score = out_norm + (size_t)N * D_DIM;

  convert_weights<<<(2 * L_DIM * D_DIM + 255) / 256, 256, 0, stream>>>(Wa, Wb, Wab);
  seg_init_k<<<1, 64, 0, stream>>>(segmax, segsum);
  fused_norm_gemm<<<N / TILE_M, 256, 0, stream>>>(feat, Wab, Wbb, ba, bb, Wc, bc,
                                                  out_norm, score_raw);
  const int eb = (N + 255) / 256;
  seg_max_k   <<<eb, 256, 0, stream>>>(score_raw, batch, segmax, N);
  seg_expsum_k<<<eb, 256, 0, stream>>>(score_raw, batch, segmax, ex, segsum, N);
  seg_div_k   <<<eb, 256, 0, stream>>>(ex, batch, segsum, out_score, N);
}
